// ExtendedS5SSM_1039382085719
// MI455X (gfx1250) — matrix operand order repaired and checked by reading, not compiled, not run on hardware
//
#include <hip/hip_runtime.h>
#include <hip/hip_bf16.h>
#include <stdint.h>

typedef __bf16 bf16;
typedef __attribute__((ext_vector_type(16))) __bf16 v16bf;
typedef __attribute__((ext_vector_type(8)))  __bf16 v8bf;
typedef __attribute__((ext_vector_type(8)))  float  v8f;

static constexpr int L_SEQ  = 8192;
static constexpr int H_DIM  = 1024;
static constexpr int P_DIM  = 512;
static constexpr int R_DIM  = 256;
static constexpr int NCHUNK = 64;
static constexpr int CLEN   = L_SEQ / NCHUNK;   // 128

// ------------------------------------------------------------------ convert
__global__ void k_f32_to_bf16(const float* __restrict__ in, bf16* __restrict__ out, int n) {
  for (int i = blockIdx.x * blockDim.x + threadIdx.x; i < n; i += gridDim.x * blockDim.x)
    out[i] = (bf16)in[i];
}

// ------------------------------------------------------- WMMA bf16 TN GEMM
// C[M,N] = alpha * A[M,K] @ W[N,K]^T  (+ C if betaFlag). Out f32 or bf16.
// 128x128 block tile, 8 waves, each wave 32(M) x 64(N) = 2x4 WMMA tiles.
// Double-buffered LDS filled with async global->LDS DMA (ASYNCcnt).
static constexpr int BM = 128, BN = 128, BK = 32;
static constexpr int LDS_STRIDE = 40;            // padded row stride (bf16 elems)

__global__ __launch_bounds__(256) void k_gemm_bf16_tn(
    const bf16* __restrict__ A, const bf16* __restrict__ W, void* __restrict__ Cv,
    int M, int N, int K, float alpha, int betaFlag, int outBf16)
{
  __shared__ bf16 As[2][BM * LDS_STRIDE];   // 2 x 10240 B
  __shared__ bf16 Bs[2][BN * LDS_STRIDE];   // 2 x 10240 B

  const int tid  = threadIdx.x;
  const int lane = tid & 31;
  const int wave = tid >> 5;        // 0..7
  const int wm   = wave & 3;        // M offset wm*32
  const int wn   = wave >> 2;       // N offset wn*64
  const int half = lane >> 4;       // 0/1
  const int l16  = lane & 15;
  const int bm0  = blockIdx.y * BM;
  const int bn0  = blockIdx.x * BN;

  v8f acc[2][4] = {};

  // async fill of one K-tile into LDS buffer `buf` (4 x b128 per thread)
  auto issue_tile = [&](int k0, int buf) {
    #pragma unroll
    for (int i = 0; i < 2; ++i) {
      int c = tid + i * 256;                    // 0..511
      int row = c >> 2, c8 = c & 3;
      unsigned dA = (unsigned)(uintptr_t)&As[buf][row * LDS_STRIDE + c8 * 8];
      const bf16* sA = A + (size_t)(bm0 + row) * K + k0 + c8 * 8;
      asm volatile("global_load_async_to_lds_b128 %0, %1, off"
                   :: "v"(dA), "v"(sA) : "memory");
      unsigned dB = (unsigned)(uintptr_t)&Bs[buf][row * LDS_STRIDE + c8 * 8];
      const bf16* sB = W + (size_t)(bn0 + row) * K + k0 + c8 * 8;
      asm volatile("global_load_async_to_lds_b128 %0, %1, off"
                   :: "v"(dB), "v"(sB) : "memory");
    }
  };

  issue_tile(0, 0);
  asm volatile("s_wait_asynccnt 0" ::: "memory");
  __syncthreads();

  int cur = 0;
  for (int k0 = 0; k0 < K; k0 += BK) {
    if (k0 + BK < K) issue_tile(k0 + BK, cur ^ 1);   // prefetch next tile (async DMA)

    // Build WMMA fragments per the CDNA5 16-bit layouts.
    union Frag { v16bf v; v8bf h[2]; };
    Frag af[2], bfrag[4];
    #pragma unroll
    for (int tm = 0; tm < 2; ++tm) {
      // A 16x32: lane<16 -> K 0..7 & 16..23; lane>=16 -> K 8..15 & 24..31
      int r = (wm * 32 + tm * 16 + l16) * LDS_STRIDE + half * 8;
      af[tm].h[0] = *(const v8bf*)&As[cur][r];
      af[tm].h[1] = *(const v8bf*)&As[cur][r + 16];
    }
    #pragma unroll
    for (int tn = 0; tn < 4; ++tn) {
      // B 32x16: lane<16 -> K 0..7 & 16..23; lane>=16 -> K 8..15 & 24..31 (N=l16)
      int r = (wn * 64 + tn * 16 + l16) * LDS_STRIDE + half * 8;
      bfrag[tn].h[0] = *(const v8bf*)&Bs[cur][r];
      bfrag[tn].h[1] = *(const v8bf*)&Bs[cur][r + 16];
    }
    #pragma unroll
    for (int tm = 0; tm < 2; ++tm)
      #pragma unroll
      for (int tn = 0; tn < 4; ++tn)
        acc[tm][tn] = __builtin_amdgcn_wmma_f32_16x16x32_bf16(
            false, af[tm].v, false, bfrag[tn].v, (short)0, acc[tm][tn], false, false);

    asm volatile("s_wait_asynccnt 0" ::: "memory");  // next tile resident
    __syncthreads();                                  // everyone done reading cur
    cur ^= 1;
  }

  // C/D layout: VGPR i -> M=i (lanes 0-15) / M=i+8 (lanes 16-31), N=l16
  float* Cf = (float*)Cv;
  bf16*  Cb = (bf16*)Cv;
  const int rowbase = bm0 + wm * 32 + half * 8;
  const int colbase = bn0 + wn * 64 + l16;
  #pragma unroll
  for (int tm = 0; tm < 2; ++tm)
    #pragma unroll
    for (int tn = 0; tn < 4; ++tn)
      #pragma unroll
      for (int i = 0; i < 8; ++i) {
        size_t idx = (size_t)(rowbase + tm * 16 + i) * N + (colbase + tn * 16);
        float v = alpha * acc[tm][tn][i];
        if (betaFlag) v += Cf[idx];
        if (outBf16) Cb[idx] = (bf16)v; else Cf[idx] = v;
      }
}

// ------------------------------------------------------- chunked diag scan
// x_k = Lam*x_{k-1} + Bu_k (complex, per channel). 3 phases across 64 chunks.
__global__ __launch_bounds__(512) void k_scan_carry(
    const float* __restrict__ bur, const float* __restrict__ bui,
    const float* __restrict__ lre, const float* __restrict__ lim,
    float* __restrict__ cre, float* __restrict__ cim,
    float* __restrict__ lnre, float* __restrict__ lnim)
{
  const int p = threadIdx.x, ch = blockIdx.x;
  const float lr = lre[p], li = lim[p];
  float xr = 0.f, xi = 0.f;
  const size_t base = (size_t)ch * CLEN * P_DIM + p;
  #pragma unroll 4
  for (int t = 0; t < CLEN; ++t) {
    const size_t idx = base + (size_t)t * P_DIM;
    const float br = bur[idx], bi = bui[idx];
    const float nr = fmaf(lr, xr, fmaf(-li, xi, br));
    const float ni = fmaf(lr, xi, fmaf( li, xr, bi));
    xr = nr; xi = ni;
  }
  cre[ch * P_DIM + p] = xr;
  cim[ch * P_DIM + p] = xi;
  if (ch == 0) {                       // Lam^CLEN, once
    float pr = 1.f, pi = 0.f;
    for (int t = 0; t < CLEN; ++t) {
      const float nr = lr * pr - li * pi;
      const float ni = lr * pi + li * pr;
      pr = nr; pi = ni;
    }
    lnre[p] = pr; lnim[p] = pi;
  }
}

__global__ __launch_bounds__(512) void k_scan_prefix(
    const float* __restrict__ cre, const float* __restrict__ cim,
    const float* __restrict__ lnre, const float* __restrict__ lnim,
    float* __restrict__ pre, float* __restrict__ pim)
{
  const int p = threadIdx.x;
  const float lr = lnre[p], li = lnim[p];
  float xr = 0.f, xi = 0.f;
  for (int j = 0; j < NCHUNK; ++j) {
    pre[j * P_DIM + p] = xr;           // state entering chunk j
    pim[j * P_DIM + p] = xi;
    const float nr = fmaf(lr, xr, fmaf(-li, xi, cre[j * P_DIM + p]));
    const float ni = fmaf(lr, xi, fmaf( li, xr, cim[j * P_DIM + p]));
    xr = nr; xi = ni;
  }
}

__global__ __launch_bounds__(512) void k_scan_apply(
    const float* __restrict__ bur, const float* __restrict__ bui,
    const float* __restrict__ lre, const float* __restrict__ lim,
    const float* __restrict__ pre, const float* __restrict__ pim,
    bf16* __restrict__ xre, bf16* __restrict__ xim)
{
  const int p = threadIdx.x, ch = blockIdx.x;
  const float lr = lre[p], li = lim[p];
  float xr = pre[ch * P_DIM + p], xi = pim[ch * P_DIM + p];
  const size_t base = (size_t)ch * CLEN * P_DIM + p;
  #pragma unroll 4
  for (int t = 0; t < CLEN; ++t) {
    const size_t idx = base + (size_t)t * P_DIM;
    const float nr = fmaf(lr, xr, fmaf(-li, xi, bur[idx]));
    const float ni = fmaf(lr, xi, fmaf( li, xr, bui[idx]));
    xr = nr; xi = ni;
    xre[idx] = (bf16)xr;               // bf16 feed for output GEMM
    xim[idx] = (bf16)xi;
  }
}

// ------------------------------------------------------------------- driver
extern "C" void kernel_launch(void* const* d_in, const int* in_sizes, int n_in,
                              void* d_out, int out_size, void* d_ws, size_t ws_size,
                              hipStream_t stream) {
  (void)in_sizes; (void)n_in; (void)out_size; (void)ws_size;
  const float* U    = (const float*)d_in[0];
  const float* F    = (const float*)d_in[1];
  const float* Hp   = (const float*)d_in[2];
  const float* Dp   = (const float*)d_in[3];
  const float* Dr   = (const float*)d_in[4];
  const float* E    = (const float*)d_in[5];
  const float* G    = (const float*)d_in[6];
  const float* LamR = (const float*)d_in[7];
  const float* LamI = (const float*)d_in[8];
  const float* Bre  = (const float*)d_in[9];
  const float* Bim  = (const float*)d_in[10];
  const float* Cre  = (const float*)d_in[11];
  const float* Cim  = (const float*)d_in[12];
  const float* Dm   = (const float*)d_in[13];
  float* Y = (float*)d_out;

  char* w = (char*)d_ws;
  size_t off = 0;
  auto take = [&](size_t bytes) { char* p = w + off; off += (bytes + 255) & ~(size_t)255; return p; };

  bf16* Ubf    = (bf16*)take((size_t)L_SEQ * H_DIM * 2);
  bf16* Fbf    = (bf16*)take((size_t)R_DIM * H_DIM * 2);
  bf16* Hpbf   = (bf16*)take((size_t)R_DIM * H_DIM * 2);
  bf16* Dpbf   = (bf16*)take((size_t)R_DIM * R_DIM * 2);
  bf16* Drbf   = (bf16*)take((size_t)R_DIM * R_DIM * 2);
  bf16* Ebf    = (bf16*)take((size_t)P_DIM * R_DIM * 2);
  bf16* Gbf    = (bf16*)take((size_t)H_DIM * R_DIM * 2);
  bf16* Brebf  = (bf16*)take((size_t)P_DIM * H_DIM * 2);
  bf16* Bimbf  = (bf16*)take((size_t)P_DIM * H_DIM * 2);
  bf16* Crebf  = (bf16*)take((size_t)H_DIM * P_DIM * 2);
  bf16* Cimbf  = (bf16*)take((size_t)H_DIM * P_DIM * 2);
  bf16* Dbf    = (bf16*)take((size_t)H_DIM * H_DIM * 2);
  bf16* t1bf   = (bf16*)take((size_t)L_SEQ * R_DIM * 2);
  bf16* t2bf   = (bf16*)take((size_t)L_SEQ * R_DIM * 2);
  bf16* pbf    = (bf16*)take((size_t)L_SEQ * R_DIM * 2);
  bf16* rbf    = (bf16*)take((size_t)L_SEQ * R_DIM * 2);
  float* BuRe  = (float*)take((size_t)L_SEQ * P_DIM * 4);
  float* BuIm  = (float*)take((size_t)L_SEQ * P_DIM * 4);
  bf16* xsReBf = (bf16*)take((size_t)L_SEQ * P_DIM * 2);
  bf16* xsImBf = (bf16*)take((size_t)L_SEQ * P_DIM * 2);
  float* carRe = (float*)take((size_t)NCHUNK * P_DIM * 4);
  float* carIm = (float*)take((size_t)NCHUNK * P_DIM * 4);
  float* preRe = (float*)take((size_t)NCHUNK * P_DIM * 4);
  float* preIm = (float*)take((size_t)NCHUNK * P_DIM * 4);
  float* lnRe  = (float*)take((size_t)P_DIM * 4);
  float* lnIm  = (float*)take((size_t)P_DIM * 4);

  auto conv = [&](const float* s, bf16* d, size_t n) {
    int blocks = (int)((n + 255) / 256); if (blocks > 2048) blocks = 2048;
    k_f32_to_bf16<<<blocks, 256, 0, stream>>>(s, d, (int)n);
  };
  auto gemm = [&](const bf16* A, const bf16* Wt, void* C, int N, int K,
                  float alpha, int beta, int obf) {
    dim3 grid(N / BN, L_SEQ / BM);
    k_gemm_bf16_tn<<<grid, 256, 0, stream>>>(A, Wt, C, L_SEQ, N, K, alpha, beta, obf);
  };

  // precision downcast of activations + weights
  conv(U,   Ubf,   (size_t)L_SEQ * H_DIM);
  conv(F,   Fbf,   (size_t)R_DIM * H_DIM);
  conv(Hp,  Hpbf,  (size_t)R_DIM * H_DIM);
  conv(Dp,  Dpbf,  (size_t)R_DIM * R_DIM);
  conv(Dr,  Drbf,  (size_t)R_DIM * R_DIM);
  conv(E,   Ebf,   (size_t)P_DIM * R_DIM);
  conv(G,   Gbf,   (size_t)H_DIM * R_DIM);
  conv(Bre, Brebf, (size_t)P_DIM * H_DIM);
  conv(Bim, Bimbf, (size_t)P_DIM * H_DIM);
  conv(Cre, Crebf, (size_t)H_DIM * P_DIM);
  conv(Cim, Cimbf, (size_t)H_DIM * P_DIM);
  conv(Dm,  Dbf,   (size_t)H_DIM * H_DIM);

  // p/r projections:  p = (U@F^T)@Dp^T,  r = (U@Hp^T)@Dr^T
  gemm(Ubf,  Fbf,  t1bf, R_DIM, H_DIM, 1.f, 0, 1);
  gemm(t1bf, Dpbf, pbf,  R_DIM, R_DIM, 1.f, 0, 1);
  gemm(Ubf,  Hpbf, t2bf, R_DIM, H_DIM, 1.f, 0, 1);
  gemm(t2bf, Drbf, rbf,  R_DIM, R_DIM, 1.f, 0, 1);

  // driving terms: Bu_re = U@Bre^T + p@E^T ; Bu_im = U@Bim^T
  gemm(Ubf, Brebf, BuRe, P_DIM, H_DIM, 1.f, 0, 0);
  gemm(pbf, Ebf,   BuRe, P_DIM, R_DIM, 1.f, 1, 0);
  gemm(Ubf, Bimbf, BuIm, P_DIM, H_DIM, 1.f, 0, 0);

  // chunked parallel scan
  k_scan_carry <<<NCHUNK, P_DIM, 0, stream>>>(BuRe, BuIm, LamR, LamI, carRe, carIm, lnRe, lnIm);
  k_scan_prefix<<<1,      P_DIM, 0, stream>>>(carRe, carIm, lnRe, lnIm, preRe, preIm);
  k_scan_apply <<<NCHUNK, P_DIM, 0, stream>>>(BuRe, BuIm, LamR, LamI, preRe, preIm, xsReBf, xsImBf);

  // y = 2*(xs_re@Cre^T - xs_im@Cim^T) + r@G^T + U@D^T
  gemm(xsReBf, Crebf, Y, H_DIM, P_DIM,  2.f, 0, 0);
  gemm(xsImBf, Cimbf, Y, H_DIM, P_DIM, -2.f, 1, 0);
  gemm(rbf,    Gbf,   Y, H_DIM, R_DIM,  1.f, 1, 0);
  gemm(Ubf,    Dbf,   Y, H_DIM, H_DIM,  1.f, 1, 0);
}
